// DeltaNet_22488448762163
// MI455X (gfx1250) — compile-verified, matrix operand order checked by reading, not run on hardware
//
#include <hip/hip_runtime.h>
#include <cstdint>
#include <cstddef>

// ---------------- problem constants ----------------
#define B_    2
#define T_    2048
#define C_    1024
#define H_    16
#define DH_   64
#define BH_   (B_*H_)          // 32 (b,h) pairs
#define BT_   (B_*T_)          // 4096 rows
#define NQKV  3072             // q|k|v concatenated
#define NGD   64               // gate(16)|decay(16)|pad
#define CH    64               // chunk length
#define NCHUNK (T_/CH)         // 32 chunks per sequence
#define EPS_      1e-6f
#define LN_EPS_   1e-5f

typedef __bf16 v16bf __attribute__((ext_vector_type(16)));
typedef float  v8f   __attribute__((ext_vector_type(8)));

// ---- WMMA helpers (CDNA5 wave32, v_wmma_f32_16x16x32_bf16) ----
// 16-bit A 16x32 lane layout: lanes 0-15: K=0..7 (v0-3), K=16..23 (v4-7);
// lanes 16-31: K=8..15, K=24..31.  B loaded from transposed storage with the
// same per-lane pattern (lane&15 selects the row/col, lane>>4 the K half).
__device__ __forceinline__ v16bf load_frag(const __bf16* p, int hi) {
  v16bf f;
  const int k0 = hi << 3;
#pragma unroll
  for (int i = 0; i < 8; ++i) f[i]     = p[k0 + i];
#pragma unroll
  for (int i = 0; i < 8; ++i) f[8 + i] = p[16 + k0 + i];
  return f;
}

__device__ __forceinline__ v8f wmma_bf16(v16bf a, v16bf b, v8f c) {
  return __builtin_amdgcn_wmma_f32_16x16x32_bf16(false, a, false, b, (short)0, c,
                                                 false, false);
}

// ---------------- pack / convert kernels ----------------
__global__ void pack_x_bf16(const float* __restrict__ x, __bf16* __restrict__ xb, int n) {
  int i = blockIdx.x * blockDim.x + threadIdx.x;
  if (i < n) xb[i] = (__bf16)x[i];
}

__global__ void pack_wqkv(const float* __restrict__ qw, const float* __restrict__ kw,
                          const float* __restrict__ vw, __bf16* __restrict__ wt) {
  int i = blockIdx.x * blockDim.x + threadIdx.x;
  if (i >= NQKV * C_) return;
  int n = i / C_, k = i - n * C_;
  float v;
  if (n < C_)            v = qw[(size_t)k * C_ + n];
  else if (n < 2 * C_)   v = kw[(size_t)k * C_ + (n - C_)];
  else                   v = vw[(size_t)k * C_ + (n - 2 * C_)];
  wt[i] = (__bf16)v;
}

__global__ void pack_wgd(const float* __restrict__ gw, const float* __restrict__ dw,
                         __bf16* __restrict__ wt) {
  int i = blockIdx.x * blockDim.x + threadIdx.x;
  if (i >= NGD * C_) return;
  int r = i / C_, k = i - r * C_;
  float v = 0.f;
  if (r < 16)       v = gw[(size_t)k * H_ + r];
  else if (r < 32)  v = dw[(size_t)k * H_ + (r - 16)];
  wt[i] = (__bf16)v;
}

__global__ void pack_wo(const float* __restrict__ ow, __bf16* __restrict__ wt) {
  int i = blockIdx.x * blockDim.x + threadIdx.x;
  if (i >= C_ * C_) return;
  int n = i / C_, k = i - n * C_;
  wt[i] = (__bf16)ow[(size_t)k * C_ + n];
}

__global__ void conv_f32_bf16(const float* __restrict__ s, __bf16* __restrict__ d, int n) {
  int i = blockIdx.x * blockDim.x + threadIdx.x;
  if (i < n) d[i] = (__bf16)s[i];
}

// ---------------- generic bf16 WMMA GEMM:  C(MxN,f32) = A(MxK) * Bt(NxK)^T ----
// Software-pipelined: iteration k+1's fragments are loaded before iteration
// k's WMMAs issue, so waits drain only to the next iteration's load count
// instead of loadcnt==0 (one exposed load latency per K-loop, not per iter).
__global__ __launch_bounds__(256)
void gemm_bf16_nt(const __bf16* __restrict__ A, const __bf16* __restrict__ Bt,
                  float* __restrict__ Cout, int M, int N, int K) {
  const int lane = threadIdx.x & 31;
  const int wave = blockIdx.x * (blockDim.x >> 5) + (threadIdx.x >> 5);
  const int ntiles = N >> 6;
  const int total  = (M >> 4) * ntiles;
  if (wave >= total) return;
  const int mt = wave / ntiles;
  const int n0 = (wave - mt * ntiles) << 6;
  const int r  = lane & 15, hi = lane >> 4;

  v8f a0 = {}, a1 = {}, a2 = {}, a3 = {};
  const __bf16* arow = A + (size_t)(mt * 16 + r) * K;
  const __bf16* b0r  = Bt + (size_t)(n0 +  0 + r) * K;
  const __bf16* b1r  = Bt + (size_t)(n0 + 16 + r) * K;
  const __bf16* b2r  = Bt + (size_t)(n0 + 32 + r) * K;
  const __bf16* b3r  = Bt + (size_t)(n0 + 48 + r) * K;

  v16bf a  = load_frag(arow, hi);
  v16bf b0 = load_frag(b0r, hi);
  v16bf b1 = load_frag(b1r, hi);
  v16bf b2 = load_frag(b2r, hi);
  v16bf b3 = load_frag(b3r, hi);
  for (int k0 = 0; k0 < K - 32; k0 += 32) {
    v16bf an  = load_frag(arow + k0 + 32, hi);
    v16bf b0n = load_frag(b0r + k0 + 32, hi);
    v16bf b1n = load_frag(b1r + k0 + 32, hi);
    v16bf b2n = load_frag(b2r + k0 + 32, hi);
    v16bf b3n = load_frag(b3r + k0 + 32, hi);
    a0 = wmma_bf16(a, b0, a0);
    a1 = wmma_bf16(a, b1, a1);
    a2 = wmma_bf16(a, b2, a2);
    a3 = wmma_bf16(a, b3, a3);
    a = an; b0 = b0n; b1 = b1n; b2 = b2n; b3 = b3n;
  }
  a0 = wmma_bf16(a, b0, a0);
  a1 = wmma_bf16(a, b1, a1);
  a2 = wmma_bf16(a, b2, a2);
  a3 = wmma_bf16(a, b3, a3);

#pragma unroll
  for (int i = 0; i < 8; ++i) {
    const size_t row = (size_t)(mt * 16 + i + hi * 8) * N;
    const int    col = n0 + (lane & 15);
    Cout[row + col]      = a0[i];
    Cout[row + col + 16] = a1[i];
    Cout[row + col + 32] = a2[i];
    Cout[row + col + 48] = a3[i];
  }
}

// ---------------- RoPE + phi + gate/decay pointwise ----------------
__device__ __forceinline__ float phi_elu1(float x) {
  return x > 0.f ? x + 1.f : __expf(x);
}

__global__ void pointwise_rope(const float* __restrict__ QKV, const float* __restrict__ GDF,
                               const float* __restrict__ gate_b, const float* __restrict__ decay_b,
                               const float* __restrict__ decay_w0,
                               __bf16* __restrict__ phiQ, __bf16* __restrict__ phiK,
                               __bf16* __restrict__ Vb,
                               float* __restrict__ gateA, float* __restrict__ rateA) {
  int idx = blockIdx.x * blockDim.x + threadIdx.x;
  if (idx >= BT_ * H_ * 32) return;
  const int j   = idx & 31;
  const int h   = (idx >> 5) & (H_ - 1);
  const int row = idx >> 9;          // b*T + t
  const int t   = row & (T_ - 1);
  const int b   = row >> 11;

  const float* q = QKV + (size_t)row * NQKV + h * DH_;
  const float* k = q + C_;
  const float* v = q + 2 * C_;

  const float invf = __powf(10000.f, -(float)j * (1.f / 32.f));
  const float f = (float)t * invf;
  const float cs = cosf(f), sn = sinf(f);

  const size_t o = ((size_t)(b * H_ + h) * T_ + t) * DH_;

  float qe = q[2 * j], qo = q[2 * j + 1];
  float ke = k[2 * j], ko = k[2 * j + 1];
  phiQ[o + 2 * j]     = (__bf16)phi_elu1(qe * cs - qo * sn);
  phiQ[o + 2 * j + 1] = (__bf16)phi_elu1(qe * sn + qo * cs);
  phiK[o + 2 * j]     = (__bf16)phi_elu1(ke * cs - ko * sn);
  phiK[o + 2 * j + 1] = (__bf16)phi_elu1(ke * sn + ko * cs);
  Vb[o + 2 * j]       = (__bf16)v[2 * j];
  Vb[o + 2 * j + 1]   = (__bf16)v[2 * j + 1];

  if (j == 0) {
    const float gv = GDF[(size_t)row * NGD + h];
    const float dv = GDF[(size_t)row * NGD + 16 + h];
    const float g  = 1.f / (1.f + __expf(-(gv + gate_b[h])));
    const float dr = dv + decay_b[h] + decay_w0[h];
    float sp = (dr > 20.f) ? dr : log1pf(__expf(dr));
    sp = fminf(fmaxf(sp, 1e-4f), 10.f);
    const size_t go = (size_t)(b * H_ + h) * T_ + t;
    gateA[go] = g;
    rateA[go] = sp;                  // log(beta) = -rate
  }
}

// ============ chunked linear-attention scan, 3-pass (parallel) ============
// Pass 1: all 32x32 (bh,chunk) blocks in parallel -> intra-chunk WMMA work:
//   SD (masked scores), row-sums, scaled Qs, Nloc = K''^T @ V, Dloc, bc.
__global__ __launch_bounds__(256)
void scan_pass1(const __bf16* __restrict__ phiQ, const __bf16* __restrict__ phiK,
                const __bf16* __restrict__ Vb,
                const float* __restrict__ gateA, const float* __restrict__ rateA,
                __bf16* __restrict__ SDbuf, __bf16* __restrict__ Qsbuf,
                float* __restrict__ rowsumB, float* __restrict__ Nloc,
                float* __restrict__ Dloc, float* __restrict__ bcArr) {
  const int bhc = blockIdx.x;                    // bh*NCHUNK + c
  const int bh = bhc >> 5, c = bhc & (NCHUNK - 1);
  const int t0 = c * CH;
  const int tid = threadIdx.x;
  const int lane = tid & 31, w = tid >> 5;
  const int hi = lane >> 4, l15 = lane & 15;
  const size_t base = (size_t)bh * T_ * DH_;
  const size_t cb = (size_t)bhc * (CH * DH_);

  __shared__ __bf16 sQ[64][64];   // raw phiQ
  __shared__ __bf16 sK[64][64];   // raw phiK -> K''^T (in-place transpose)
  __shared__ __bf16 sVt[64][64];  // V transposed [e][s]
  __shared__ float sL[64], sg[64], srow[64];

  for (int i = tid; i < 4096; i += 256) {
    const int t = i >> 6, d = i & 63;
    const size_t g = base + (size_t)(t0 + t) * DH_ + d;
    sQ[t][d]  = phiQ[g];
    sK[t][d]  = phiK[g];
    sVt[d][t] = Vb[g];
  }
  if (tid < 64) {
    sg[tid]   = gateA[(size_t)bh * T_ + t0 + tid];
    sL[tid]   = -rateA[(size_t)bh * T_ + t0 + tid];
    srow[tid] = 0.f;
  }
  __syncthreads();
  if (tid == 0) {                                // inclusive prefix of log(beta)
    float L = 0.f;
#pragma unroll 8
    for (int t = 0; t < 64; ++t) { L += sL[t]; sL[t] = L; }
  }
  __syncthreads();
  const float Lc = sL[63];

  // ---- S = phiQ @ phiK^T, mask/scale, spill SD + rowsums ----
#pragma unroll
  for (int rep = 0; rep < 2; ++rep) {
    const int tile = w + rep * 8;
    const int tm = tile >> 2, tn = tile & 3;
    v8f acc = {};
#pragma unroll
    for (int kk = 0; kk < 2; ++kk) {
      v16bf a  = load_frag(&sQ[tm * 16 + l15][kk * 32], hi);
      v16bf bb = load_frag(&sK[tn * 16 + l15][kk * 32], hi);
      acc = wmma_bf16(a, bb, acc);
    }
#pragma unroll
    for (int i = 0; i < 8; ++i) {
      const int t = tm * 16 + i + hi * 8;
      const int s = tn * 16 + l15;
      float v = 0.f;
      if (s <= t) {
        v = acc[i] * __expf(sL[t] - sL[s]) * sg[s];
        atomicAdd(&srow[t], v);
      }
      SDbuf[cb + t * 64 + s] = (__bf16)v;
    }
  }
  __syncthreads();

  // ---- stage K for transpose; emit Qs and rowsums ----
  float kv[16];
#pragma unroll
  for (int jj = 0; jj < 16; ++jj) {
    const int i = tid * 16 + jj;
    kv[jj] = (float)sK[i >> 6][i & 63];
  }
  for (int i = tid; i < 4096; i += 256) {
    const int t = i >> 6, d = i & 63;
    Qsbuf[cb + i] = (__bf16)((float)sQ[t][d] * __expf(sL[t]));
  }
  if (tid < 64) rowsumB[(size_t)bh * T_ + t0 + tid] = srow[tid];
  __syncthreads();
#pragma unroll
  for (int jj = 0; jj < 16; ++jj) {              // sK <- K''^T
    const int i = tid * 16 + jj;
    const int s = i >> 6, d = i & 63;
    sK[d][s] = (__bf16)(kv[jj] * sg[s] * __expf(Lc - sL[s]));
  }
  __syncthreads();

  // ---- Nloc = K''^T @ V ; Dloc = rowsum(K''^T) ; bc ----
#pragma unroll
  for (int rep = 0; rep < 2; ++rep) {
    const int tile = w + rep * 8;
    const int tm = tile >> 2, tn = tile & 3;
    v8f acc = {};
#pragma unroll
    for (int kk = 0; kk < 2; ++kk) {
      v16bf a  = load_frag(&sK[tm * 16 + l15][kk * 32], hi);
      v16bf bb = load_frag(&sVt[tn * 16 + l15][kk * 32], hi);
      acc = wmma_bf16(a, bb, acc);
    }
#pragma unroll
    for (int i = 0; i < 8; ++i) {
      const int d = tm * 16 + i + hi * 8;
      const int e = tn * 16 + l15;
      Nloc[cb + d * 64 + e] = acc[i];
    }
  }
  if (tid < 64) {
    float s = 0.f;
    for (int ss = 0; ss < 64; ++ss) s += (float)sK[tid][ss];
    Dloc[(size_t)bhc * 64 + tid] = s;
  }
  if (tid == 0) bcArr[bhc] = __expf(Lc);
}

// Pass 2: per-element linear scan over chunks (no barriers, fully parallel
// across the 32*4096 state elements + 32*64 D elements).
__global__ void scan_pass2(const float* __restrict__ Nloc, const float* __restrict__ Dloc,
                           const float* __restrict__ bcArr,
                           float* __restrict__ N0g, float* __restrict__ D0g) {
  const int idx = blockIdx.x * blockDim.x + threadIdx.x;
  const int NELT = BH_ * 4096;
  if (idx < NELT) {
    const int bh = idx >> 12, e = idx & 4095;
    float n = 0.f;
#pragma unroll 4
    for (int c = 0; c < NCHUNK; ++c) {
      const size_t o = ((size_t)bh * NCHUNK + c) * 4096 + e;
      N0g[o] = n;
      n = bcArr[bh * NCHUNK + c] * n + Nloc[o];
    }
  } else if (idx < NELT + BH_ * 64) {
    const int r = idx - NELT;
    const int bh = r >> 6, d = r & 63;
    float n = 0.f;
#pragma unroll 4
    for (int c = 0; c < NCHUNK; ++c) {
      const size_t o = ((size_t)bh * NCHUNK + c) * 64 + d;
      D0g[o] = n;
      n = bcArr[bh * NCHUNK + c] * n + Dloc[o];
    }
  }
}

// Pass 3: Y = SD @ V + Qs @ N0, divide by denominator, write (b,t,c) layout.
__global__ __launch_bounds__(256)
void scan_pass3(const __bf16* __restrict__ SDbuf, const __bf16* __restrict__ Qsbuf,
                const __bf16* __restrict__ Vb, const float* __restrict__ rowsumB,
                const float* __restrict__ N0g, const float* __restrict__ D0g,
                float* __restrict__ Ybuf) {
  const int bhc = blockIdx.x;
  const int bh = bhc >> 5, c = bhc & (NCHUNK - 1);
  const int b = bh / H_, h = bh - b * H_;
  const int t0 = c * CH;
  const int tid = threadIdx.x, lane = tid & 31, w = tid >> 5;
  const int hi = lane >> 4, l15 = lane & 15;
  const size_t cb = (size_t)bhc * (CH * DH_);
  const size_t base = (size_t)bh * T_ * DH_;

  __shared__ __bf16 sQs[64][64];  // scaled Q [t][d]
  __shared__ __bf16 sSD[64][64];  // masked scores [t][s]
  __shared__ __bf16 sVt[64][64];  // V transposed [e][s]
  __shared__ __bf16 sNb[64][64];  // N0 transposed bf16 [e][d]
  __shared__ float sD0[64], sden[64];

  for (int i = tid; i < 4096; i += 256) {
    const int rr = i >> 6, q = i & 63;
    sQs[rr][q] = Qsbuf[cb + i];
    sSD[rr][q] = SDbuf[cb + i];
    sVt[q][rr] = Vb[base + (size_t)(t0 + rr) * DH_ + q];
    sNb[q][rr] = (__bf16)N0g[cb + i];      // N0[d=rr][e=q] -> sNb[e][d]
  }
  if (tid < 64) sD0[tid] = D0g[(size_t)bhc * 64 + tid];
  __syncthreads();
  if (tid < 64) {
    float s = 0.f;
    for (int d = 0; d < 64; ++d) s += (float)sQs[tid][d] * sD0[d];
    sden[tid] = rowsumB[(size_t)bh * T_ + t0 + tid] + s + EPS_;
  }
  __syncthreads();

#pragma unroll
  for (int rep = 0; rep < 2; ++rep) {
    const int tile = w + rep * 8;
    const int tm = tile >> 2, tn = tile & 3;
    v8f acc = {};
#pragma unroll
    for (int kk = 0; kk < 2; ++kk) {
      v16bf a  = load_frag(&sSD[tm * 16 + l15][kk * 32], hi);
      v16bf bb = load_frag(&sVt[tn * 16 + l15][kk * 32], hi);
      acc = wmma_bf16(a, bb, acc);
    }
#pragma unroll
    for (int kk = 0; kk < 2; ++kk) {
      v16bf a  = load_frag(&sQs[tm * 16 + l15][kk * 32], hi);
      v16bf bb = load_frag(&sNb[tn * 16 + l15][kk * 32], hi);
      acc = wmma_bf16(a, bb, acc);
    }
#pragma unroll
    for (int i = 0; i < 8; ++i) {
      const int t = tm * 16 + i + hi * 8;
      const int e = tn * 16 + l15;
      Ybuf[(size_t)(b * T_ + t0 + t) * C_ + h * DH_ + e] = acc[i] / sden[t];
    }
  }
}

// ---------------- residual + bias + LayerNorm ----------------
__global__ __launch_bounds__(256)
void ln_kernel(const float* __restrict__ x, const float* __restrict__ Obuf,
               const float* __restrict__ out_b, const float* __restrict__ ln_g,
               const float* __restrict__ ln_b, float* __restrict__ out) {
  const int row = blockIdx.x;
  float vals[4], s = 0.f, s2 = 0.f;
#pragma unroll
  for (int i = 0; i < 4; ++i) {
    const int c = threadIdx.x + i * 256;
    const float v = x[(size_t)row * C_ + c] + Obuf[(size_t)row * C_ + c] + out_b[c];
    vals[i] = v; s += v; s2 += v * v;
  }
  __shared__ float r1[256], r2[256];
  r1[threadIdx.x] = s; r2[threadIdx.x] = s2;
  __syncthreads();
  for (int off = 128; off > 0; off >>= 1) {
    if (threadIdx.x < off) {
      r1[threadIdx.x] += r1[threadIdx.x + off];
      r2[threadIdx.x] += r2[threadIdx.x + off];
    }
    __syncthreads();
  }
  const float mu  = r1[0] * (1.f / C_);
  const float var = r2[0] * (1.f / C_) - mu * mu;
  const float rs  = rsqrtf(var + LN_EPS_);
#pragma unroll
  for (int i = 0; i < 4; ++i) {
    const int c = threadIdx.x + i * 256;
    out[(size_t)row * C_ + c] = (vals[i] - mu) * rs * ln_g[c] + ln_b[c];
  }
}

// ---------------- host-side orchestration ----------------
static inline size_t alnup(size_t x) { return (x + 255) & ~(size_t)255; }

extern "C" void kernel_launch(void* const* d_in, const int* in_sizes, int n_in,
                              void* d_out, int out_size, void* d_ws, size_t ws_size,
                              hipStream_t stream) {
  (void)in_sizes; (void)n_in; (void)out_size; (void)ws_size;
  const float* x        = (const float*)d_in[0];
  const float* q_w      = (const float*)d_in[1];
  const float* k_w      = (const float*)d_in[2];
  const float* v_w      = (const float*)d_in[3];
  const float* out_w    = (const float*)d_in[4];
  const float* out_b    = (const float*)d_in[5];
  const float* decay_w  = (const float*)d_in[6];
  const float* decay_b  = (const float*)d_in[7];
  const float* gate_w   = (const float*)d_in[8];
  const float* gate_b   = (const float*)d_in[9];
  const float* decay_w0 = (const float*)d_in[10];
  const float* ln_g     = (const float*)d_in[11];
  const float* ln_b     = (const float*)d_in[12];
  float* out = (float*)d_out;

  char* p = (char*)d_ws;
  size_t off = 0;
  auto take = [&](size_t bytes) { char* q = p + off; off = alnup(off + bytes); return (void*)q; };

  __bf16* Xbf    = (__bf16*)take((size_t)BT_ * C_ * 2);
  __bf16* WqkvT  = (__bf16*)take((size_t)NQKV * C_ * 2);
  __bf16* WgdT   = (__bf16*)take((size_t)NGD * C_ * 2);
  __bf16* WoT    = (__bf16*)take((size_t)C_ * C_ * 2);
  float*  QKVf   = (float*) take((size_t)BT_ * NQKV * 4);
  float*  GDF    = (float*) take((size_t)BT_ * NGD * 4);
  __bf16* phiQ   = (__bf16*)take((size_t)BT_ * C_ * 2);
  __bf16* phiK   = (__bf16*)take((size_t)BT_ * C_ * 2);
  __bf16* Vbf    = (__bf16*)take((size_t)BT_ * C_ * 2);
  float*  gateA  = (float*) take((size_t)BT_ * H_ * 4);
  float*  rateA  = (float*) take((size_t)BT_ * H_ * 4);
  __bf16* SDbuf  = (__bf16*)take((size_t)BH_ * NCHUNK * CH * DH_ * 2);
  __bf16* Qsbuf  = (__bf16*)take((size_t)BH_ * NCHUNK * CH * DH_ * 2);
  float*  rowsumB= (float*) take((size_t)BH_ * T_ * 4);
  float*  Nloc   = (float*) take((size_t)BH_ * NCHUNK * DH_ * DH_ * 4);
  float*  Dloc   = (float*) take((size_t)BH_ * NCHUNK * DH_ * 4);
  float*  bcArr  = (float*) take((size_t)BH_ * NCHUNK * 4);
  float*  N0g    = (float*) take((size_t)BH_ * NCHUNK * DH_ * DH_ * 4);
  float*  D0g    = (float*) take((size_t)BH_ * NCHUNK * DH_ * 4);
  float*  Ybuf   = (float*) take((size_t)BT_ * C_ * 4);
  __bf16* Ybf    = (__bf16*)take((size_t)BT_ * C_ * 2);
  float*  Obuf   = (float*) take((size_t)BT_ * C_ * 4);

  const int TPB = 256;
  // pack / convert
  pack_x_bf16<<<(BT_ * C_ + TPB - 1) / TPB, TPB, 0, stream>>>(x, Xbf, BT_ * C_);
  pack_wqkv  <<<(NQKV * C_ + TPB - 1) / TPB, TPB, 0, stream>>>(q_w, k_w, v_w, WqkvT);
  pack_wgd   <<<(NGD * C_ + TPB - 1) / TPB, TPB, 0, stream>>>(gate_w, decay_w, WgdT);
  pack_wo    <<<(C_ * C_ + TPB - 1) / TPB, TPB, 0, stream>>>(out_w, WoT);

  // projections (bf16 WMMA, fp32 accumulate)
  {
    int waves = (BT_ / 16) * (NQKV / 64);
    gemm_bf16_nt<<<waves / 8, TPB, 0, stream>>>(Xbf, WqkvT, QKVf, BT_, NQKV, C_);
  }
  {
    int waves = (BT_ / 16) * (NGD / 64);
    gemm_bf16_nt<<<(waves + 7) / 8, TPB, 0, stream>>>(Xbf, WgdT, GDF, BT_, NGD, C_);
  }

  // RoPE + phi + gate/decay
  pointwise_rope<<<(BT_ * H_ * 32 + TPB - 1) / TPB, TPB, 0, stream>>>(
      QKVf, GDF, gate_b, decay_b, decay_w0, phiQ, phiK, Vbf, gateA, rateA);

  // 3-pass chunked scan: 1024-way parallel intra-chunk WMMA work, tiny
  // elementwise inter-chunk scan, then 1024-way parallel output pass.
  scan_pass1<<<BH_ * NCHUNK, TPB, 0, stream>>>(phiQ, phiK, Vbf, gateA, rateA,
                                               SDbuf, Qsbuf, rowsumB, Nloc, Dloc, bcArr);
  {
    int nthr = BH_ * 4096 + BH_ * 64;
    scan_pass2<<<(nthr + TPB - 1) / TPB, TPB, 0, stream>>>(Nloc, Dloc, bcArr, N0g, D0g);
  }
  scan_pass3<<<BH_ * NCHUNK, TPB, 0, stream>>>(SDbuf, Qsbuf, Vbf, rowsumB, N0g, D0g, Ybuf);

  // output projection + residual + LN
  conv_f32_bf16<<<(BT_ * C_ + TPB - 1) / TPB, TPB, 0, stream>>>(Ybuf, Ybf, BT_ * C_);
  {
    int waves = (BT_ / 16) * (C_ / 64);
    gemm_bf16_nt<<<waves / 8, TPB, 0, stream>>>(Ybf, WoT, Obuf, BT_, C_, C_);
  }
  ln_kernel<<<BT_, TPB, 0, stream>>>(x, Obuf, out_b, ln_g, ln_b, out);
}
